// MSG_67259187855617
// MI455X (gfx1250) — matrix operand order repaired and checked by reading, not compiled, not run on hardware
//
// CDNA5 / gfx1250 implementation of the multi-scale graph-hypergraph network.
//
// Roofline reasoning (MI455X): dominant work is batched An(256x256) @ H(256xL)
// propagation (~38 GFLOP, ~63 FLOP/byte -> compute bound) plus ~7 GFLOP of 1x1
// channel contractions. All of these are routed through ONE generic batched
// WMMA kernel using v_wmma_f32_16x16x32_f16 (f16 operands, f32 accumulate;
// adjacency weights are row-normalized <=1 and activations O(1), so f16 range
// is safe). Total HBM traffic ~1 GB -> ~45us at 23.3 TB/s.
//
// Round-1 fix: the boundary handling in the WMMA kernel used per-lane
// predicated loads, which lowered to exec-mask save/restore + s_wait_loadcnt 0
// around EVERY element load. Since an out-of-range B *column* only pollutes an
// output column that is never stored, we clamp the column address and load
// unconditionally; K-loop is unrolled x2 with two operand sets so loads of the
// next chunk are in flight while the current WMMA executes.

#include <hip/hip_runtime.h>
#include <hip/hip_bf16.h>
#include <math.h>

typedef __attribute__((ext_vector_type(16))) _Float16 v16h;
typedef __attribute__((ext_vector_type(8)))  float    v8f;

#define B_    8
#define N_    256
#define T_    168
#define CC_   32
#define SC_   16
#define EC_   128
#define OD_   12
#define ND_   40
#define KK_   32     // hyperedges
#define NLAY_ 3

// Load one 16x32 A tile (f16 row-major, lane holds k = 16*half + j) and one 32x16
// B tile (f32 -> f16 gather, columns across lanes). No predication.
__device__ __forceinline__ void wmma_load_ab(const _Float16* __restrict__ arow,
                                             const float* __restrict__ bp,
                                             long Lc, v16h& a, v16h& b)
{
#pragma unroll
  for (int j = 0; j < 16; ++j) a[j] = arow[j];
#pragma unroll
  for (int j = 0; j < 16; ++j) b[j] = (_Float16)bp[(long)j * Lc];
}

// ---------------------------------------------------------------------------
// Generic batched WMMA matmul:
//   Out[batch, v, l] (+)= alpha * ( sum_k A[v,k]*In[batch,k,l] + bias ) +
//                         beta * X[batch, v, l]
// A is f16 row-major with row stride strideA (halfs). M = gridDim.y*16.
// In batch stride = K*Lc, Out/X batch stride = M*Lc.
// bias_mode: 0 none, 1 bias[batch & 31], 2 bias[v].
// One wave (32 lanes) per 16x16 output tile; K stepped by 32, unrolled x2.
// ---------------------------------------------------------------------------
__global__ __launch_bounds__(32) void k_wmma_bmm(
    float* __restrict__ Out, const float* __restrict__ In,
    const _Float16* __restrict__ A, int strideA,
    const float* __restrict__ X, const float* __restrict__ bias,
    int K, int Lc, float alpha, float beta, int bias_mode, int accumulate)
{
  const int lane = threadIdx.x;
  const int col  = lane & 15;     // column within tile / A row within tile
  const int half = lane >> 4;     // K-half selector
  const int l0   = blockIdx.x * 16;
  const int v0   = blockIdx.y * 16;
  const int batch = blockIdx.z;
  const int M    = gridDim.y * 16;

  const int  lcol  = l0 + col;
  const bool colOK = (lcol < Lc);
  const int  lsafe = colOK ? lcol : (Lc - 1);   // clamp: garbage cols never stored

  // A-operand lane layout (16-bit A 16x32): lane m holds row v0+m;
  // halfs 0..15 = K(kk+half*16 ..+15), matching B
  const _Float16* arow = A + (long)(v0 + col) * (long)strideA + half * 16;
  // B-operand (32x16): lanes 0-15 = cols, K kk..kk+15; lanes 16-31 K kk+16..+31
  const float* bp = In + (long)batch * K * Lc + (long)(half * 16) * Lc + lsafe;
  const long bstep = (long)32 * Lc;

  v8f acc = {};
  int kk = 0;
  for (; kk + 64 <= K; kk += 64) {
    v16h a0, b0, a1, b1;
    wmma_load_ab(arow,        bp,         Lc, a0, b0);
    wmma_load_ab(arow + 32,   bp + bstep, Lc, a1, b1);
    acc = __builtin_amdgcn_wmma_f32_16x16x32_f16(false, a0, false, b0,
                                                 (short)0, acc, false, false);
    acc = __builtin_amdgcn_wmma_f32_16x16x32_f16(false, a1, false, b1,
                                                 (short)0, acc, false, false);
    arow += 64; bp += 2 * bstep;
  }
  if (kk < K) {
    v16h a0, b0;
    wmma_load_ab(arow, bp, Lc, a0, b0);
    acc = __builtin_amdgcn_wmma_f32_16x16x32_f16(false, a0, false, b0,
                                                 (short)0, acc, false, false);
  }

  if (!colOK) return;
  const int rbase = half * 8;   // C/D layout: lanes 16-31 hold rows M=8..15
  float bb = 0.0f;
  if (bias_mode == 1) bb = bias[batch & (CC_ - 1)];
  const long outB = (long)batch * M * Lc;
#pragma unroll
  for (int r = 0; r < 8; ++r) {
    const int  v   = v0 + rbase + r;
    const long idx = outB + (long)v * Lc + lcol;
    float bval = (bias_mode == 2) ? bias[v] : bb;
    float val  = alpha * (acc[r] + bval);
    if (beta != 0.0f) val += beta * X[idx];
    if (accumulate)   val += Out[idx];
    Out[idx] = val;
  }
}

// --------------------------- small prep kernels ----------------------------

// Fold the 1x1 start conv (CIN=1) into the first temporal conv's weights.
__global__ void k_prep_w0(const float* __restrict__ ws, const float* __restrict__ bs,
                          const float* __restrict__ w0, const float* __restrict__ b0,
                          float* __restrict__ w0eff, float* __restrict__ b0eff)
{
  int co = threadIdx.x;                 // 32
  float be = b0[co];
  for (int k = 0; k < 7; ++k) {
    float acc = 0.f;
    for (int ci = 0; ci < 32; ++ci) acc += w0[(co * 32 + ci) * 7 + k] * ws[ci];
    w0eff[co * 7 + k] = acc;
  }
  for (int ci = 0; ci < 32; ++ci) {
    float sw = 0.f;
    for (int k = 0; k < 7; ++k) sw += w0[(co * 32 + ci) * 7 + k];
    be += sw * bs[ci];
  }
  b0eff[co] = be;
}

// dst[o*Cin + c] = (f16) src[c*Cout + o]   (transpose weights for WMMA A)
__global__ void k_t2h(const float* __restrict__ src, _Float16* __restrict__ dst,
                      int Cin, int Cout)
{
  int i = blockIdx.x * blockDim.x + threadIdx.x;
  if (i < Cin * Cout) {
    int o = i / Cin, c = i % Cin;
    dst[o * Cin + c] = (_Float16)src[c * Cout + o];
  }
}

// --------------------------- temporal convolutions -------------------------

__global__ void k_conv0(const float* __restrict__ x, const float* __restrict__ w0eff,
                        const float* __restrict__ b0eff, float* __restrict__ s0)
{
  __shared__ float wk[7];
  __shared__ float bsh;
  int n = blockIdx.x, co = blockIdx.y, b = blockIdx.z;
  int to = threadIdx.x;
  if (to < 7) wk[to] = w0eff[co * 7 + to];
  if (to == 0) bsh = b0eff[co];
  __syncthreads();
  if (to < 162) {
    const float* xp = x + ((long)b * N_ + n) * T_ + to;
    float acc = bsh;
#pragma unroll
    for (int k = 0; k < 7; ++k) acc += wk[k] * xp[k];
    s0[(((long)(b * CC_ + co)) * N_ + n) * 162 + to] = acc;
  }
}

// generic stride-2 temporal conv, CC->CC
__global__ void k_convs(const float* __restrict__ sin, const float* __restrict__ w,
                        const float* __restrict__ bias, float* __restrict__ sout,
                        int Lin, int Lout, int Kk)
{
  extern __shared__ float wl[];   // 32*Kk
  int n = blockIdx.x, co = blockIdx.y, b = blockIdx.z;
  int t = threadIdx.x;
  for (int i = t; i < 32 * Kk; i += blockDim.x) wl[i] = w[co * 32 * Kk + i];
  __syncthreads();
  if (t < Lout) {
    float acc = bias[co];
    for (int ci = 0; ci < 32; ++ci) {
      const float* sp = sin + ((long)(b * CC_ + ci) * N_ + n) * Lin + t * 2;
      const float* wp = wl + ci * Kk;
      for (int k = 0; k < Kk; ++k) acc += wp[k] * sp[k];
    }
    sout[((long)(b * CC_ + co) * N_ + n) * Lout + t] = acc;
  }
}

// scale0: full-T conv of raw input -> outsc[b][0][sc][n]
__global__ void k_scale0(const float* __restrict__ x, const float* __restrict__ w,
                         const float* __restrict__ bias, float* __restrict__ outsc)
{
  int n = blockIdx.x, b = blockIdx.y, sc = threadIdx.x;   // 16
  const float* xp = x + ((long)b * N_ + n) * T_;
  const float* wp = w + sc * T_;
  float acc = bias[sc];
  for (int t = 0; t < T_; ++t) acc += wp[t] * xp[t];
  outsc[(((long)b * 4 + 0) * SC_ + sc) * N_ + n] = acc;
}

// --------------------------- graph constructor -----------------------------

__global__ void k_xfeat(const float* __restrict__ x, const float* __restrict__ gcw,
                        const float* __restrict__ gcb, float* __restrict__ xfeat)
{
  __shared__ float xm[T_];
  int n = blockIdx.x;
  for (int t = threadIdx.x; t < T_; t += blockDim.x) {
    float s = 0.f;
    for (int b = 0; b < B_; ++b) s += x[((long)b * N_ + n) * T_ + t];
    xm[t] = s * (1.0f / B_);
  }
  __syncthreads();
  int d = threadIdx.x;
  if (d < ND_) {
    float s = gcb[d];
    for (int t = 0; t < T_; ++t) s += xm[t] * gcw[t * ND_ + d];
    xfeat[n * ND_ + d] = tanhf(s);
  }
}

__global__ void k_nv_init(const float* __restrict__ emb1, const float* __restrict__ emb2,
                          const float* __restrict__ xfeat,
                          float* __restrict__ nv1, float* __restrict__ nv2)
{
  int i = blockIdx.x * blockDim.x + threadIdx.x;
  if (i < N_ * ND_) {
    float xf = xfeat[i];
    nv1[i] = emb1[i] + xf;
    nv2[i] = emb2[i] + xf;
  }
}

__global__ void k_nv_layer(float* __restrict__ nv, const float* __restrict__ W,
                           const float* __restrict__ bvec, float scale)
{
  __shared__ float row[ND_];
  int n = blockIdx.x, d = threadIdx.x;     // 64 threads
  if (d < ND_) row[d] = nv[n * ND_ + d] * scale;
  __syncthreads();
  if (d < ND_) {
    float s = bvec[d];
    for (int e = 0; e < ND_; ++e) s += row[e] * W[e * ND_ + d];
    nv[n * ND_ + d] = tanhf(3.0f * s);
  }
}

__global__ void k_adj(const float* __restrict__ nv1, const float* __restrict__ nv2,
                      float* __restrict__ adj0)
{
  __shared__ float r1[ND_], r2[ND_];
  int v = blockIdx.x, w = threadIdx.x;     // 256 threads
  if (w < ND_) { r1[w] = nv1[v * ND_ + w]; r2[w] = nv2[v * ND_ + w]; }
  __syncthreads();
  float a = 0.f;
  for (int e = 0; e < ND_; ++e)
    a += r1[e] * nv2[w * ND_ + e] - r2[e] * nv1[w * ND_ + e];
  float t = tanhf(3.0f * a);
  adj0[v * N_ + w] = t > 0.f ? t : 0.f;
}

// 20th-largest value per row -> threshold (ties beyond 20 are zeros -> no-op)
__global__ void k_topk_thr(const float* __restrict__ adj0, float* __restrict__ thr)
{
  int r = blockIdx.x * blockDim.x + threadIdx.x;   // 256 total
  if (r >= N_) return;
  float best[20];
  for (int i = 0; i < 20; ++i) best[i] = -1.f;     // values are >= 0
  for (int w = 0; w < N_; ++w) {
    float v = adj0[r * N_ + w];
    if (v > best[19]) {
      int p = 19;
      while (p > 0 && best[p - 1] < v) { best[p] = best[p - 1]; --p; }
      best[p] = v;
    }
  }
  thr[r] = best[19];
}

// Build both row-normalized adjacencies (forward + transpose) directly as f16.
__global__ void k_an_build(const float* __restrict__ adj0, const float* __restrict__ thr,
                           _Float16* __restrict__ An1h, _Float16* __restrict__ An2h)
{
  __shared__ float s1[N_], s2[N_];
  int v = blockIdx.x, w = threadIdx.x;   // 256 threads
  float a1 = adj0[v * N_ + w];
  float a2 = adj0[w * N_ + v];
  float m1 = (a1 >= thr[v]) ? a1 : 0.f;
  float m2 = (a2 >= thr[w]) ? a2 : 0.f;
  if (w == v) { m1 += 1.f; m2 += 1.f; }
  s1[w] = m1; s2[w] = m2;
  __syncthreads();
  for (int s = 128; s > 0; s >>= 1) {
    if (w < s) { s1[w] += s1[w + s]; s2[w] += s2[w + s]; }
    __syncthreads();
  }
  float inv1 = 1.f / s1[0], inv2 = 1.f / s2[0];
  An1h[v * N_ + w] = (_Float16)(m1 * inv1);
  An2h[v * N_ + w] = (_Float16)(m2 * inv2);
}

// --------------------------- hypergraph construction -----------------------

__global__ void k_smean(const float* __restrict__ s, float* __restrict__ smean, int L)
{
  int n = blockIdx.x, c = threadIdx.x;   // 32
  float acc = 0.f;
  for (int b = 0; b < B_; ++b) {
    const float* p = s + ((long)(b * CC_ + c) * N_ + n) * L;
    for (int l = 0; l < L; ++l) acc += p[l];
  }
  smean[n * CC_ + c] = acc / (float)(B_ * L);
}

__global__ void k_hbuild(const float* __restrict__ smean, const float* __restrict__ fw,
                         const float* __restrict__ fb, const float* __restrict__ ee,
                         float* __restrict__ H)
{
  __shared__ float sm[CC_];
  __shared__ float nf[ND_];
  int n = blockIdx.x, t = threadIdx.x;   // 64
  if (t < CC_) sm[t] = smean[n * CC_ + t];
  __syncthreads();
  if (t < ND_) {
    float s = fb[t];
    for (int c = 0; c < CC_; ++c) s += sm[c] * fw[c * ND_ + t];
    nf[t] = tanhf(s);
  }
  __syncthreads();
  if (t < KK_) {
    float s = 0.f;
    for (int d = 0; d < ND_; ++d) s += nf[d] * ee[t * ND_ + d];
    float th = tanhf(3.0f * s);
    H[n * KK_ + t] = th > 0.f ? th : 0.f;
  }
}

// HdeT[k][n] = H[n][k] / (sum_n H + 1e-6)   (f16, A-operand for node->edge)
__global__ void k_de(const float* __restrict__ H, _Float16* __restrict__ HdeT)
{
  int k = threadIdx.x;                   // 32
  float s = 1e-6f;
  for (int n = 0; n < N_; ++n) s += H[n * KK_ + k];
  float inv = 1.f / s;
  for (int n = 0; n < N_; ++n) HdeT[k * N_ + n] = (_Float16)(H[n * KK_ + k] * inv);
}

// Hn[n][k] = H[n][k] / (sum_k H + 1e-6)     (f16, A-operand for edge->node)
__global__ void k_dv(const float* __restrict__ H, _Float16* __restrict__ Hnh)
{
  int n = threadIdx.x;                   // 256
  float s = 1e-6f;
  for (int k = 0; k < KK_; ++k) s += H[n * KK_ + k];
  float inv = 1.f / s;
  for (int k = 0; k < KK_; ++k) Hnh[n * KK_ + k] = (_Float16)(H[n * KK_ + k] * inv);
}

// --------------------------- output head -----------------------------------

// outsc[b][i][sc][n] = sum_{c,l} g[b,c,n,l]*scw[sc,c,l] + scb[sc]
__global__ void k_scale_out(const float* __restrict__ g, const float* __restrict__ w,
                            const float* __restrict__ bias, float* __restrict__ outsc,
                            int L, int scale_idx)
{
  __shared__ float red[64];
  int n = blockIdx.x, sc = blockIdx.y, b = blockIdx.z;
  int t = threadIdx.x;                   // 64
  float acc = 0.f;
  for (int c = 0; c < CC_; ++c) {
    const float* gp = g + ((long)(b * CC_ + c) * N_ + n) * L;
    const float* wp = w + (sc * CC_ + c) * L;
    for (int l = t; l < L; l += 64) acc += gp[l] * wp[l];
  }
  red[t] = acc; __syncthreads();
  for (int s = 32; s > 0; s >>= 1) { if (t < s) red[t] += red[t + s]; __syncthreads(); }
  if (t == 0)
    outsc[(((long)b * 4 + scale_idx) * SC_ + sc) * N_ + n] = red[0] + bias[sc];
}

__global__ void k_se(const float* __restrict__ outsc, const float* __restrict__ w1,
                     const float* __restrict__ w2, float* __restrict__ se2)
{
  __shared__ float se[64];
  __shared__ float r[4];
  int b = blockIdx.x, j = threadIdx.x;   // 64
  float acc = 0.f;
  const float* p = outsc + ((long)b * 64 + j) * N_;
  for (int n = 0; n < N_; ++n) acc += p[n];
  se[j] = acc * (1.0f / N_);
  __syncthreads();
  if (j < 4) {
    float s = 0.f;
    for (int k = 0; k < 64; ++k) s += se[k] * w1[k * 4 + j];
    r[j] = s > 0.f ? s : 0.f;
  }
  __syncthreads();
  if (j < 4) {
    float s = 0.f;
    for (int k = 0; k < 4; ++k) s += r[k] * w2[k * 4 + j];
    se2[b * 4 + j] = 1.0f / (1.0f + expf(-s));
  }
}

__global__ void k_final(const float* __restrict__ outsc, const float* __restrict__ se2,
                        const float* __restrict__ e1w, const float* __restrict__ e1b,
                        const float* __restrict__ e2w, const float* __restrict__ e2b,
                        float* __restrict__ out)
{
  __shared__ float y[SC_];
  __shared__ float t1[EC_];
  int n = blockIdx.x, b = blockIdx.y, t = threadIdx.x;   // 128
  if (t < SC_) {
    float f = 0.f;
    for (int i = 0; i < 4; ++i)
      f += outsc[(((long)b * 4 + i) * SC_ + t) * N_ + n] * se2[b * 4 + i];
    f *= 0.25f;
    y[t] = f > 0.f ? f : 0.f;
  }
  __syncthreads();
  {
    float s = e1b[t];
    for (int sc = 0; sc < SC_; ++sc) s += y[sc] * e1w[sc * EC_ + t];
    t1[t] = s > 0.f ? s : 0.f;
  }
  __syncthreads();
  if (t < OD_) {
    float s = e2b[t];
    for (int e = 0; e < EC_; ++e) s += t1[e] * e2w[e * OD_ + t];
    out[((long)b * OD_ + t) * N_ + n] = s;
  }
}

// ---------------------------------------------------------------------------

static inline void bmm(hipStream_t st, float* Out, const float* In,
                       const _Float16* A, int strideA,
                       const float* X, const float* bias,
                       int M, int K, int Lc, float alpha, float beta,
                       int bias_mode, int acc, int batch)
{
  dim3 g((unsigned)((Lc + 15) / 16), (unsigned)(M / 16), (unsigned)batch);
  k_wmma_bmm<<<g, 32, 0, st>>>(Out, In, A, strideA, X, bias, K, Lc,
                               alpha, beta, bias_mode, acc);
}

extern "C" void kernel_launch(void* const* d_in, const int* in_sizes, int n_in,
                              void* d_out, int out_size, void* d_ws, size_t ws_size,
                              hipStream_t stream)
{
  (void)in_sizes; (void)n_in; (void)out_size; (void)ws_size;

  const float* x          = (const float*)d_in[0];
  const float* sc_start_w = (const float*)d_in[1];
  const float* sc_start_b = (const float*)d_in[2];
  const float* sc_w0      = (const float*)d_in[3];
  const float* sc_b0      = (const float*)d_in[4];
  const float* sc_w1      = (const float*)d_in[5];
  const float* sc_b1      = (const float*)d_in[6];
  const float* sc_w2      = (const float*)d_in[7];
  const float* sc_b2      = (const float*)d_in[8];
  const float* scale0_w   = (const float*)d_in[9];
  const float* scale0_b   = (const float*)d_in[10];
  const float* emb1       = (const float*)d_in[11];
  const float* emb2       = (const float*)d_in[12];
  const float* gc_lin1_w  = (const float*)d_in[13];
  const float* gc_lin1_b  = (const float*)d_in[14];
  const float* gc_lin2_w  = (const float*)d_in[15];
  const float* gc_lin2_b  = (const float*)d_in[16];
  const float* gc_x_w     = (const float*)d_in[17];
  const float* gc_x_b     = (const float*)d_in[18];
  const float* hc_feat_w  = (const float*)d_in[19];
  const float* hc_feat_b  = (const float*)d_in[20];
  const float* hc_edge    = (const float*)d_in[21];
  const float* mix1_w     = (const float*)d_in[22];
  const float* mix1_b     = (const float*)d_in[23];
  const float* mix2_w     = (const float*)d_in[24];
  const float* mix2_b     = (const float*)d_in[25];
  const float* hyp_w      = (const float*)d_in[26];
  const float* hyp_b      = (const float*)d_in[27];
  const float* scw[3]     = {(const float*)d_in[28], (const float*)d_in[30], (const float*)d_in[32]};
  const float* scb[3]     = {(const float*)d_in[29], (const float*)d_in[31], (const float*)d_in[33]};
  const float* gf_w1      = (const float*)d_in[34];
  const float* gf_w2      = (const float*)d_in[35];
  const float* end1_w     = (const float*)d_in[36];
  const float* end1_b     = (const float*)d_in[37];
  const float* end2_w     = (const float*)d_in[38];
  const float* end2_b     = (const float*)d_in[39];
  float* out = (float*)d_out;

  const int LENS[3] = {162, 79, 39};
  const float SCALES[3] = {1.0f, 0.8f, 0.6f};

  // ---- workspace bump allocator (256B aligned) ----
  char* base = (char*)d_ws;
  size_t off = 0;
  auto alloc_f = [&](size_t n) -> float* {
    float* p = (float*)(base + off);
    off = (off + n * sizeof(float) + 255) & ~(size_t)255;
    return p;
  };
  auto alloc_h = [&](size_t n) -> _Float16* {
    _Float16* p = (_Float16*)(base + off);
    off = (off + n * sizeof(_Float16) + 255) & ~(size_t)255;
    return p;
  };

  float* s0 = alloc_f((size_t)B_ * CC_ * N_ * 162);
  float* s1 = alloc_f((size_t)B_ * CC_ * N_ * 79);
  float* s2 = alloc_f((size_t)B_ * CC_ * N_ * 39);
  float* scales_buf[3] = {s0, s1, s2};
  float* h1 = alloc_f((size_t)B_ * CC_ * N_ * 162);
  float* h2 = alloc_f((size_t)B_ * CC_ * N_ * 162);
  float* g  = alloc_f((size_t)B_ * CC_ * N_ * 162);
  float* eb = alloc_f((size_t)B_ * CC_ * KK_ * 162);
  float* fb = alloc_f((size_t)B_ * CC_ * KK_ * 162);
  float* adj0  = alloc_f((size_t)N_ * N_);
  float* thr   = alloc_f(N_);
  float* xfeat = alloc_f((size_t)N_ * ND_);
  float* nv1   = alloc_f((size_t)N_ * ND_);
  float* nv2   = alloc_f((size_t)N_ * ND_);
  float* smean = alloc_f((size_t)N_ * CC_);
  float* Hbuf  = alloc_f((size_t)N_ * KK_);
  float* w0eff = alloc_f(CC_ * 7);
  float* b0eff = alloc_f(CC_);
  float* outsc = alloc_f((size_t)B_ * 4 * SC_ * N_);
  float* se2   = alloc_f(B_ * 4);
  _Float16* An1h   = alloc_h((size_t)NLAY_ * N_ * N_);
  _Float16* An2h   = alloc_h((size_t)NLAY_ * N_ * N_);
  _Float16* HdeT_h = alloc_h((size_t)KK_ * N_);
  _Float16* Hn_h   = alloc_h((size_t)N_ * KK_);
  _Float16* mix1T  = alloc_h((size_t)CC_ * 96);
  _Float16* mix2T  = alloc_h((size_t)CC_ * 96);
  _Float16* thetaT = alloc_h((size_t)CC_ * CC_);

  // ---- temporal convs (start 1x1 conv folded into conv0 weights) ----
  k_prep_w0<<<1, 32, 0, stream>>>(sc_start_w, sc_start_b, sc_w0, sc_b0, w0eff, b0eff);
  k_conv0<<<dim3(N_, CC_, B_), 176, 0, stream>>>(x, w0eff, b0eff, s0);
  k_convs<<<dim3(N_, CC_, B_), 96, 32 * 6 * sizeof(float), stream>>>(s0, sc_w1, sc_b1, s1, 162, 79, 6);
  k_convs<<<dim3(N_, CC_, B_), 96, 32 * 3 * sizeof(float), stream>>>(s1, sc_w2, sc_b2, s2, 79, 39, 3);

  // ---- dynamic graph constructor ----
  k_xfeat<<<N_, 64, 0, stream>>>(x, gc_x_w, gc_x_b, xfeat);
  k_nv_init<<<(N_ * ND_ + 255) / 256, 256, 0, stream>>>(emb1, emb2, xfeat, nv1, nv2);
  for (int i = 0; i < NLAY_; ++i) {
    k_nv_layer<<<N_, 64, 0, stream>>>(nv1, gc_lin1_w + i * ND_ * ND_, gc_lin1_b + i * ND_, SCALES[i]);
    k_nv_layer<<<N_, 64, 0, stream>>>(nv2, gc_lin2_w + i * ND_ * ND_, gc_lin2_b + i * ND_, SCALES[i]);
    k_adj<<<N_, N_, 0, stream>>>(nv1, nv2, adj0);
    k_topk_thr<<<8, 32, 0, stream>>>(adj0, thr);
    k_an_build<<<N_, N_, 0, stream>>>(adj0, thr,
                                      An1h + (size_t)i * N_ * N_,
                                      An2h + (size_t)i * N_ * N_);
  }

  // ---- scale0 head ----
  k_scale0<<<dim3(N_, B_), SC_, 0, stream>>>(x, scale0_w, scale0_b, outsc);

  // ---- per-scale graph + hypergraph propagation ----
  for (int i = 0; i < NLAY_; ++i) {
    const int L = LENS[i];
    const int NL = N_ * L;
    float* s = scales_buf[i];
    const _Float16* A1 = An1h + (size_t)i * N_ * N_;
    const _Float16* A2 = An2h + (size_t)i * N_ * N_;

    // hypergraph incidence from this scale
    k_smean<<<N_, CC_, 0, stream>>>(s, smean, L);
    k_hbuild<<<N_, 64, 0, stream>>>(smean, hc_feat_w + i * CC_ * ND_,
                                    hc_feat_b + i * ND_, hc_edge + i * KK_ * ND_, Hbuf);
    k_de<<<1, KK_, 0, stream>>>(Hbuf, HdeT_h);
    k_dv<<<1, N_, 0, stream>>>(Hbuf, Hn_h);

    // layer weights as f16 transposed A-operands
    k_t2h<<<(96 * CC_ + 255) / 256, 256, 0, stream>>>(mix1_w + i * 96 * CC_, mix1T, 96, CC_);
    k_t2h<<<(96 * CC_ + 255) / 256, 256, 0, stream>>>(mix2_w + i * 96 * CC_, mix2T, 96, CC_);
    k_t2h<<<(CC_ * CC_ + 255) / 256, 256, 0, stream>>>(hyp_w + i * CC_ * CC_, thetaT, CC_, CC_);

    // mixprop forward direction: h1 = 0.05 s + 0.95 An1 @ s ; h2 likewise
    bmm(stream, h1, s,  A1, N_, s, nullptr, N_, N_, L, 0.95f, 0.05f, 0, 0, B_ * CC_);
    bmm(stream, h2, h1, A1, N_, s, nullptr, N_, N_, L, 0.95f, 0.05f, 0, 0, B_ * CC_);
    // pconv(cat(s,h1,h2), mix1) * 0.6 -> g  (K=96 split into 3 K=32 slices)
    bmm(stream, g, s,  mix1T + 0,  96, nullptr, nullptr,        CC_, CC_, NL, 0.6f, 0.f, 0, 0, B_);
    bmm(stream, g, h1, mix1T + 32, 96, nullptr, nullptr,        CC_, CC_, NL, 0.6f, 0.f, 0, 1, B_);
    bmm(stream, g, h2, mix1T + 64, 96, nullptr, mix1_b + i*CC_, CC_, CC_, NL, 0.6f, 0.f, 2, 1, B_);

    // mixprop transpose direction
    bmm(stream, h1, s,  A2, N_, s, nullptr, N_, N_, L, 0.95f, 0.05f, 0, 0, B_ * CC_);
    bmm(stream, h2, h1, A2, N_, s, nullptr, N_, N_, L, 0.95f, 0.05f, 0, 0, B_ * CC_);
    bmm(stream, g, s,  mix2T + 0,  96, nullptr, nullptr,        CC_, CC_, NL, 0.6f, 0.f, 0, 1, B_);
    bmm(stream, g, h1, mix2T + 32, 96, nullptr, nullptr,        CC_, CC_, NL, 0.6f, 0.f, 0, 1, B_);
    bmm(stream, g, h2, mix2T + 64, 96, nullptr, mix2_b + i*CC_, CC_, CC_, NL, 0.6f, 0.f, 2, 1, B_);

    // hypergraph: e = (H/De)^T @ s ; f = theta^T applied over channels of e ;
    // g += 0.4 * ( (H/Dv) @ f + hyp_b )
    bmm(stream, eb, s,  HdeT_h, N_,  nullptr, nullptr,       KK_, N_,  L,       1.0f, 0.f, 0, 0, B_ * CC_);
    bmm(stream, fb, eb, thetaT, CC_, nullptr, nullptr,       CC_, CC_, KK_ * L, 1.0f, 0.f, 0, 0, B_);
    bmm(stream, g,  fb, Hn_h,   KK_, nullptr, hyp_b + i*CC_, N_,  KK_, L,       0.4f, 0.f, 1, 1, B_ * CC_);

    // project to scale channels with full-L temporal reduction
    k_scale_out<<<dim3(N_, SC_, B_), 64, 0, stream>>>(g, scw[i], scb[i], outsc, L, i + 1);
  }

  // ---- SE-gated fusion + end MLP ----
  k_se<<<B_, 64, 0, stream>>>(outsc, gf_w1, gf_w2, se2);
  k_final<<<dim3(N_, B_), EC_, 0, stream>>>(outsc, se2, end1_w, end1_b, end2_w, end2_b, out);
}
